// MultiScaleAttention_22368189677632
// MI455X (gfx1250) — compile-verified, matrix operand order checked by reading, not run on hardware
//
#include <hip/hip_runtime.h>
#include <hip/hip_bf16.h>
#include <cstdint>

// ---------------------------------------------------------------------------
// MViTv2 MultiScaleAttention for gfx1250 (MI455X), wave32 + WMMA bf16.
// All WMMA B/A operands staged in LDS in *fragment-packed* order so every
// fragment read is two aligned ds_load_b128 (no per-element ds_load_u16).
// ---------------------------------------------------------------------------

typedef __attribute__((ext_vector_type(16))) __bf16 v16bf;
typedef __attribute__((ext_vector_type(8)))  float  v8f;

#define NUM_HEADS 8
#define HEAD_DIM  96
#define D_MODEL   768
#define B_SZ      4
#define NQ        3137          // 56*56 + 1
#define NK        785           // 28*28 + 1
#define SPQ       3136
#define KW        28
#define M_ROWS    (B_SZ * NQ)   // 12548
#define SCALE_F   0.10206207261596577f
#define EPS_F     1e-6f

// Fragment element mapping (16x16x32 bf16): lane-half kh holds K chunks
// {kh*8..kh*8+7} and {16+kh*8..16+kh*8+7}; element e indexes VGPR halves.
__device__ __forceinline__ int kmap(int e, int kh) {
  return (e < 8) ? (kh * 8 + e) : (8 + kh * 8 + e);
}
// Inverse: K value -> (kh, e) position inside a fragment.
__device__ __forceinline__ void kpos(int k, int& kh, int& e) {
  if (k < 16) { kh = k >> 3; e = k & 7; }
  else        { kh = (k - 16) >> 3; e = 8 + ((k - 16) & 7); }
}

// Row-major A-fragment load using 4x float4 (global_load_b128): the two
// K-chunks {kh*8..} and {16+kh*8..} are contiguous 32B runs.
__device__ __forceinline__ v16bf load_frag_rm(const float* __restrict__ p, int kh,
                                              float scale) {
  const float4 a0 = *(const float4*)(p + kh * 8);
  const float4 a1 = *(const float4*)(p + kh * 8 + 4);
  const float4 a2 = *(const float4*)(p + 16 + kh * 8);
  const float4 a3 = *(const float4*)(p + 16 + kh * 8 + 4);
  v16bf f;
  f[0]  = (__bf16)(a0.x * scale); f[1]  = (__bf16)(a0.y * scale);
  f[2]  = (__bf16)(a0.z * scale); f[3]  = (__bf16)(a0.w * scale);
  f[4]  = (__bf16)(a1.x * scale); f[5]  = (__bf16)(a1.y * scale);
  f[6]  = (__bf16)(a1.z * scale); f[7]  = (__bf16)(a1.w * scale);
  f[8]  = (__bf16)(a2.x * scale); f[9]  = (__bf16)(a2.y * scale);
  f[10] = (__bf16)(a2.z * scale); f[11] = (__bf16)(a2.w * scale);
  f[12] = (__bf16)(a3.x * scale); f[13] = (__bf16)(a3.y * scale);
  f[14] = (__bf16)(a3.z * scale); f[15] = (__bf16)(a3.w * scale);
  return f;
}

// Fragment-packed LDS read: slot = (tile*32 + lane)*16, 32B aligned -> 2x b128.
__device__ __forceinline__ v16bf load_frag_packed(const __bf16* __restrict__ base,
                                                  int slot) {
  return *(const v16bf*)(base + (size_t)slot * 16);
}

// Stage a K-row (k = krow_in_tile, fixed per lane) of an f32 tile into
// fragment-packed LDS: value for column c of tile t goes to
// lds[(t*32 + c + 16*khw)*16 + ew].  Four float4 per 16-column tile.
__device__ __forceinline__ void stage_row_packed(const float* __restrict__ src,
                                                 __bf16* __restrict__ lds,
                                                 int khw, int ew, int ntiles) {
#pragma unroll
  for (int t = 0; t < 6; ++t) {
    if (t >= ntiles) break;
#pragma unroll
    for (int u = 0; u < 4; ++u) {
      float4 val = *(const float4*)(src + t * 16 + u * 4);
      __bf16* d = lds + (size_t)(t * 32 + u * 4 + 16 * khw) * 16 + ew;
      d[0]  = (__bf16)val.x;
      d[16] = (__bf16)val.y;
      d[32] = (__bf16)val.z;
      d[48] = (__bf16)val.w;
    }
  }
}

__device__ __forceinline__ float redmax16(float x) {
#pragma unroll
  for (int o = 1; o < 16; o <<= 1) x = fmaxf(x, __shfl_xor(x, o, 32));
  return x;
}
__device__ __forceinline__ float redsum16(float x) {
#pragma unroll
  for (int o = 1; o < 16; o <<= 1) x += __shfl_xor(x, o, 32);
  return x;
}

// ---------------------------------------------------------------------------
// Kernel 1: qkv = x @ w_qkv + b, scattered into q/k/v (B,H,NQ,96) buffers.
// One wave per block; 16x64 output tile; B fragment-packed in LDS.
// ---------------------------------------------------------------------------
__global__ __launch_bounds__(32) void qkv_gemm_kernel(
    const float* __restrict__ x, const float* __restrict__ w,
    const float* __restrict__ bias, float* __restrict__ qd,
    float* __restrict__ kd, float* __restrict__ vd) {
  const int lane = threadIdx.x & 31, lo = lane & 15, hi = lane >> 4;
  const int n0 = blockIdx.x * 64, m0 = blockIdx.y * 16;
  int row = m0 + lo;
  if (row >= M_ROWS) row = M_ROWS - 1;
  int khw, ew;
  kpos(lane, khw, ew);

  __shared__ __align__(32) __bf16 btile[4 * 32 * 16];  // 4 tiles, packed

  v8f acc[4];
#pragma unroll
  for (int t = 0; t < 4; ++t)
#pragma unroll
    for (int i = 0; i < 8; ++i) acc[t][i] = 0.f;

  const float* xrow = x + (size_t)row * D_MODEL;
  for (int k0 = 0; k0 < D_MODEL; k0 += 32) {
    stage_row_packed(w + (size_t)(k0 + lane) * 2304 + n0, btile, khw, ew, 4);
    __syncthreads();
    v16bf a = load_frag_rm(xrow + k0, hi, 1.0f);
#pragma unroll
    for (int t = 0; t < 4; ++t) {
      v16bf b = load_frag_packed(btile, t * 32 + lane);
      acc[t] = __builtin_amdgcn_wmma_f32_16x16x32_bf16(false, a, false, b,
                                                       (short)0, acc[t], false, false);
    }
    __syncthreads();
  }

#pragma unroll
  for (int t = 0; t < 4; ++t) {
    const int j = n0 + t * 16 + lo;
    const int which = j / D_MODEL;
    const int rem = j - which * D_MODEL;
    const int h = rem / HEAD_DIM, c = rem - h * HEAD_DIM;
    float* dst = (which == 0) ? qd : (which == 1) ? kd : vd;
    const float bj = bias[j];
#pragma unroll
    for (int i = 0; i < 8; ++i) {
      int r = m0 + i + 8 * hi;
      if (r >= M_ROWS) continue;
      int bb = r / NQ, n = r - bb * NQ;
      dst[(((size_t)bb * NUM_HEADS + h) * NQ + n) * HEAD_DIM + c] = acc[t][i] + bj;
    }
  }
}

// ---------------------------------------------------------------------------
// Kernel 2: depthwise 3x3 conv (pad 1, stride s) on spatial tokens + LayerNorm.
// ---------------------------------------------------------------------------
__global__ __launch_bounds__(HEAD_DIM) void pool_ln_kernel(
    const float* __restrict__ in, const float* __restrict__ wc,
    const float* __restrict__ g, const float* __restrict__ be,
    float* __restrict__ out, int stride, int out_hw) {
  const int ntok = out_hw * out_hw + 1;
  const int bh = blockIdx.x / ntok;
  const int tok = blockIdx.x - bh * ntok;
  const int c = threadIdx.x;
  const float* src = in + (size_t)bh * NQ * HEAD_DIM;

  float val;
  if (tok == 0) {
    val = src[c];
  } else {
    const int sp = tok - 1;
    const int oy = sp / out_hw, ox = sp - oy * out_hw;
    float s = 0.f;
#pragma unroll
    for (int dy = 0; dy < 3; ++dy) {
      int iy = oy * stride - 1 + dy;
      if (iy < 0 || iy >= 56) continue;
#pragma unroll
      for (int dx = 0; dx < 3; ++dx) {
        int ix = ox * stride - 1 + dx;
        if (ix < 0 || ix >= 56) continue;
        s += wc[(dy * 3 + dx) * HEAD_DIM + c] *
             src[(size_t)(1 + iy * 56 + ix) * HEAD_DIM + c];
      }
    }
    val = s;
  }

  __shared__ float red[HEAD_DIM];
  red[c] = val;
  __syncthreads();
  float m = 0.f;
  for (int i = 0; i < HEAD_DIM; ++i) m += red[i];
  m *= (1.f / HEAD_DIM);
  float vv = 0.f;
  for (int i = 0; i < HEAD_DIM; ++i) {
    float d = red[i] - m;
    vv += d * d;
  }
  vv *= (1.f / HEAD_DIM);
  out[((size_t)bh * ntok + tok) * HEAD_DIM + c] =
      (val - m) * rsqrtf(vv + EPS_F) * g[c] + be[c];
}

// ---------------------------------------------------------------------------
// Kernel 3: rel-pos bias einsums (float4 vectorized dot over 96 channels).
// ---------------------------------------------------------------------------
__global__ void relpos_kernel(const float* __restrict__ qp,
                              const float* __restrict__ rph,
                              const float* __restrict__ rpw,
                              float* __restrict__ rel_h,
                              float* __restrict__ rel_w) {
  const long long total = (long long)B_SZ * NUM_HEADS * SPQ * KW;
  long long idx = (long long)blockIdx.x * blockDim.x + threadIdx.x;
  if (idx >= total) return;
  int kk = (int)(idx % KW);
  int sp = (int)((idx / KW) % SPQ);
  int bh = (int)(idx / ((long long)KW * SPQ));
  int y = sp / 56, xx = sp - y * 56;
  const float4* q4 = (const float4*)(qp + ((size_t)bh * NQ + 1 + sp) * HEAD_DIM);
  const float4* rh4 = (const float4*)(rph + (size_t)(y - 2 * kk + 54) * HEAD_DIM);
  const float4* rw4 = (const float4*)(rpw + (size_t)(xx - 2 * kk + 54) * HEAD_DIM);
  float sh = 0.f, sw = 0.f;
#pragma unroll 6
  for (int c = 0; c < HEAD_DIM / 4; ++c) {
    float4 a = q4[c], hh = rh4[c], ww = rw4[c];
    sh += a.x * hh.x + a.y * hh.y + a.z * hh.z + a.w * hh.w;
    sw += a.x * ww.x + a.y * ww.y + a.z * ww.z + a.w * ww.w;
  }
  rel_h[idx] = sh;
  rel_w[idx] = sw;
}

// ---------------------------------------------------------------------------
// Kernel 4: flash attention per (b,h), 16 query rows per wave/block.
// V tiles and P staged fragment-packed in LDS; bias tables staged in LDS.
// ---------------------------------------------------------------------------
__global__ __launch_bounds__(32) void flash_attn_kernel(
    const float* __restrict__ qp, const float* __restrict__ kp,
    const float* __restrict__ vp, const float* __restrict__ rel_h,
    const float* __restrict__ rel_w, float* __restrict__ ctx) {
  const int lane = threadIdx.x & 31, lo = lane & 15, hi = lane >> 4;
  const int q0 = blockIdx.x * 16;
  const int bh = blockIdx.y;
  const int b = bh >> 3, h = bh & 7;
  int khw, ew;
  kpos(lane, khw, ew);

  const float* qbase = qp + (size_t)bh * NQ * HEAD_DIM;
  const float* kbase = kp + (size_t)bh * NK * HEAD_DIM;
  const float* vbase = vp + (size_t)bh * NK * HEAD_DIM;

  __shared__ float bh_s[16 * KW], bw_s[16 * KW];
  __shared__ __align__(32) __bf16 pbuf[32 * 16];       // A-frag packed P
  __shared__ __align__(32) __bf16 vtile[6 * 32 * 16];  // 6 packed channel tiles

  // Stage this q-tile's rel-pos bias rows once (rows q0..q0+15).
  {
    const float* rhb = rel_h + (size_t)bh * SPQ * KW;
    const float* rwb = rel_w + (size_t)bh * SPQ * KW;
    for (int idx = lane; idx < 16 * KW; idx += 32) {
      int r = idx / KW, kk = idx - r * KW;
      int qr = q0 + r;
      float vh = 0.f, vw = 0.f;
      if (qr >= 1 && qr < NQ) {
        vh = rhb[(size_t)(qr - 1) * KW + kk];
        vw = rwb[(size_t)(qr - 1) * KW + kk];
      }
      bh_s[idx] = vh;
      bw_s[idx] = vw;
    }
  }

  // q fragments for the whole row-tile (K = 96 split into 3 chunks of 32).
  int qrow_l = q0 + lo;
  if (qrow_l >= NQ) qrow_l = NQ - 1;
  v16bf qa[3];
#pragma unroll
  for (int kc = 0; kc < 3; ++kc)
    qa[kc] = load_frag_rm(qbase + (size_t)qrow_l * HEAD_DIM + kc * 32, hi, SCALE_F);

  v8f acc[6];
#pragma unroll
  for (int t = 0; t < 6; ++t)
#pragma unroll
    for (int i = 0; i < 8; ++i) acc[t][i] = 0.f;

  float mrow[8], lrow[8];
#pragma unroll
  for (int i = 0; i < 8; ++i) {
    mrow[i] = -1e30f;
    lrow[i] = 0.f;
  }
  __syncthreads();  // bias tables ready

  for (int kb = 0; kb < 800; kb += 32) {  // 25 steps of 32 keys (785 padded)
    // Stage V[kb+lane, :] into 6 fragment-packed channel tiles.
    {
      int vrow = kb + lane;
      if (vrow >= NK) vrow = NK - 1;
      stage_row_packed(vbase + (size_t)vrow * HEAD_DIM, vtile, khw, ew, 6);
    }

    v8f S[2];
#pragma unroll
    for (int t = 0; t < 2; ++t) {
#pragma unroll
      for (int i = 0; i < 8; ++i) S[t][i] = 0.f;
      int krow = kb + t * 16 + lo;
      if (krow >= NK) krow = NK - 1;
#pragma unroll
      for (int kc = 0; kc < 3; ++kc) {
        v16bf kf = load_frag_rm(kbase + (size_t)krow * HEAD_DIM + kc * 32, hi, 1.0f);
        S[t] = __builtin_amdgcn_wmma_f32_16x16x32_bf16(false, qa[kc], false, kf,
                                                       (short)0, S[t], false, false);
      }
    }

    // rel-pos bias (from LDS) + masking, tile-local row max.
    float tmax[8];
#pragma unroll
    for (int i = 0; i < 8; ++i) tmax[i] = -1e30f;
#pragma unroll
    for (int t = 0; t < 2; ++t) {
      const int key = kb + t * 16 + lo;
#pragma unroll
      for (int i = 0; i < 8; ++i) {
        const int qr = q0 + i + 8 * hi;
        const int r = i + 8 * hi;
        float s = S[t][i];
        if (key >= NK || qr >= NQ) {
          s = -1e30f;
        } else if (qr >= 1 && key >= 1) {
          const int kk = key - 1;
          s += bh_s[r * KW + (kk / KW)] + bw_s[r * KW + (kk % KW)];
        }
        S[t][i] = s;
        tmax[i] = fmaxf(tmax[i], s);
      }
    }
#pragma unroll
    for (int i = 0; i < 8; ++i) tmax[i] = redmax16(tmax[i]);

    // Online-softmax state update; write P directly in A-fragment order.
    float rs[8];
#pragma unroll
    for (int i = 0; i < 8; ++i) {
      float mn = fmaxf(mrow[i], tmax[i]);
      rs[i] = __expf(mrow[i] - mn);
      mrow[i] = mn;
    }
    float psum[8];
#pragma unroll
    for (int i = 0; i < 8; ++i) psum[i] = 0.f;
#pragma unroll
    for (int t = 0; t < 2; ++t) {
      const int c32 = t * 16 + lo;     // key column within the 32-key step
      int khd, ed;
      kpos(c32, khd, ed);
#pragma unroll
      for (int i = 0; i < 8; ++i) {
        float p = __expf(S[t][i] - mrow[i]);
        psum[i] += p;
        pbuf[(size_t)((i + 8 * hi) + 16 * khd) * 16 + ed] = (__bf16)p;
      }
    }
#pragma unroll
    for (int i = 0; i < 8; ++i) lrow[i] = lrow[i] * rs[i] + redsum16(psum[i]);
    __syncthreads();  // pbuf + vtile visible

    // P A-fragment: one aligned 32B read (2x ds_load_b128).
    v16bf pa = load_frag_packed(pbuf, lane);

    // Rescale accumulators and do O += P @ V across the 6 channel tiles.
#pragma unroll
    for (int t = 0; t < 6; ++t) {
#pragma unroll
      for (int i = 0; i < 8; ++i) acc[t][i] *= rs[i];
      v16bf vf = load_frag_packed(vtile, t * 32 + lane);
      acc[t] = __builtin_amdgcn_wmma_f32_16x16x32_bf16(false, pa, false, vf,
                                                       (short)0, acc[t], false, false);
    }
    __syncthreads();  // protect vtile/pbuf against next iteration's writes
  }

  // Normalize, add pooled-q residual, gather heads into (B, NQ, 768).
#pragma unroll
  for (int t = 0; t < 6; ++t) {
    const int c = t * 16 + lo;
#pragma unroll
    for (int i = 0; i < 8; ++i) {
      const int qr = q0 + i + 8 * hi;
      if (qr < NQ) {
        float val = acc[t][i] / lrow[i] + qbase[(size_t)qr * HEAD_DIM + c];
        ctx[((size_t)b * NQ + qr) * D_MODEL + h * HEAD_DIM + c] = val;
      }
    }
  }
}

// ---------------------------------------------------------------------------
// Kernel 5: out = ctx @ w_proj + b_proj.  16x64 per-wave tiles, packed LDS B.
// ---------------------------------------------------------------------------
__global__ __launch_bounds__(32) void proj_gemm_kernel(
    const float* __restrict__ ctx, const float* __restrict__ w,
    const float* __restrict__ bias, float* __restrict__ out) {
  const int lane = threadIdx.x & 31, lo = lane & 15, hi = lane >> 4;
  const int n0 = blockIdx.x * 64, m0 = blockIdx.y * 16;
  int row = m0 + lo;
  if (row >= M_ROWS) row = M_ROWS - 1;
  int khw, ew;
  kpos(lane, khw, ew);

  __shared__ __align__(32) __bf16 btile[4 * 32 * 16];

  v8f acc[4];
#pragma unroll
  for (int t = 0; t < 4; ++t)
#pragma unroll
    for (int i = 0; i < 8; ++i) acc[t][i] = 0.f;

  const float* crow = ctx + (size_t)row * D_MODEL;
  for (int k0 = 0; k0 < D_MODEL; k0 += 32) {
    stage_row_packed(w + (size_t)(k0 + lane) * D_MODEL + n0, btile, khw, ew, 4);
    __syncthreads();
    v16bf a = load_frag_rm(crow + k0, hi, 1.0f);
#pragma unroll
    for (int t = 0; t < 4; ++t) {
      v16bf b = load_frag_packed(btile, t * 32 + lane);
      acc[t] = __builtin_amdgcn_wmma_f32_16x16x32_bf16(false, a, false, b,
                                                       (short)0, acc[t], false, false);
    }
    __syncthreads();
  }

#pragma unroll
  for (int t = 0; t < 4; ++t) {
    const float bj = bias[n0 + t * 16 + lo];
#pragma unroll
    for (int i = 0; i < 8; ++i) {
      int r = m0 + i + 8 * hi;
      if (r >= M_ROWS) continue;
      out[(size_t)r * D_MODEL + n0 + t * 16 + lo] = acc[t][i] + bj;
    }
  }
}

// ---------------------------------------------------------------------------
extern "C" void kernel_launch(void* const* d_in, const int* in_sizes, int n_in,
                              void* d_out, int out_size, void* d_ws, size_t ws_size,
                              hipStream_t stream) {
  (void)in_sizes; (void)n_in; (void)out_size; (void)ws_size;

  const float* x       = (const float*)d_in[0];
  const float* w_qkv   = (const float*)d_in[1];
  const float* b_qkv   = (const float*)d_in[2];
  const float* conv_q  = (const float*)d_in[3];
  const float* conv_k  = (const float*)d_in[4];
  const float* conv_v  = (const float*)d_in[5];
  const float* gq      = (const float*)d_in[6];
  const float* bq      = (const float*)d_in[7];
  const float* gk      = (const float*)d_in[8];
  const float* bk      = (const float*)d_in[9];
  const float* gv      = (const float*)d_in[10];
  const float* bv      = (const float*)d_in[11];
  const float* rel_ph  = (const float*)d_in[12];
  const float* rel_pw  = (const float*)d_in[13];
  const float* w_proj  = (const float*)d_in[14];
  const float* b_proj  = (const float*)d_in[15];
  float* out = (float*)d_out;

  // Workspace partition (floats).
  const size_t QS = (size_t)B_SZ * NUM_HEADS * NQ * HEAD_DIM;  // 9,636,864
  const size_t KS = (size_t)B_SZ * NUM_HEADS * NK * HEAD_DIM;  // 2,411,520
  const size_t RS = (size_t)B_SZ * NUM_HEADS * SPQ * KW;       // 2,809,856
  float* ws   = (float*)d_ws;
  float* q_raw = ws;
  float* k_raw = q_raw + QS;
  float* v_raw = k_raw + QS;
  float* q_p   = v_raw + QS;
  float* k_p   = q_p + QS;
  float* v_p   = k_p + KS;
  float* relh  = v_p + KS;
  float* relw  = relh + RS;
  float* ctx   = relw + RS;   // M_ROWS * D_MODEL floats

  const int mtiles = (M_ROWS + 15) / 16;  // 785

  // 1) fused qkv projection + head split (N = 2304 -> 36 column blocks of 64)
  qkv_gemm_kernel<<<dim3(3 * D_MODEL / 64, mtiles), 32, 0, stream>>>(
      x, w_qkv, b_qkv, q_raw, k_raw, v_raw);

  // 2) attention pooling + layernorm
  pool_ln_kernel<<<B_SZ * NUM_HEADS * NQ, HEAD_DIM, 0, stream>>>(
      q_raw, conv_q, gq, bq, q_p, 1, 56);
  pool_ln_kernel<<<B_SZ * NUM_HEADS * NK, HEAD_DIM, 0, stream>>>(
      k_raw, conv_k, gk, bk, k_p, 2, 28);
  pool_ln_kernel<<<B_SZ * NUM_HEADS * NK, HEAD_DIM, 0, stream>>>(
      v_raw, conv_v, gv, bv, v_p, 2, 28);

  // 3) decomposed rel-pos bias tables
  {
    long long total = (long long)B_SZ * NUM_HEADS * SPQ * KW;
    int blocks = (int)((total + 255) / 256);
    relpos_kernel<<<blocks, 256, 0, stream>>>(q_p, rel_ph, rel_pw, relh, relw);
  }

  // 4) flash attention + residual + head gather
  flash_attn_kernel<<<dim3((NQ + 15) / 16, B_SZ * NUM_HEADS), 32, 0, stream>>>(
      q_p, k_p, v_p, relh, relw, ctx);

  // 5) output projection (N = 768 -> 12 column blocks of 64)
  proj_gemm_kernel<<<dim3(D_MODEL / 64, mtiles), 32, 0, stream>>>(
      ctx, w_proj, b_proj, out);
}
